// CustomAttention_5068061409377
// MI455X (gfx1250) — matrix operand order repaired and checked by reading, not compiled, not run on hardware
//
#include <hip/hip_runtime.h>

// ---------------------------------------------------------------------------
// Types
// ---------------------------------------------------------------------------
typedef __bf16 bf16;
typedef __attribute__((ext_vector_type(16))) __bf16 bf16x16;
typedef __attribute__((ext_vector_type(8)))  __bf16 bf16x8;
typedef __attribute__((ext_vector_type(8)))  float  f32x8;
typedef __attribute__((ext_vector_type(4)))  float  f32x4;
typedef __attribute__((ext_vector_type(4)))  int    i32x4;

// Pointer-to-global / pointer-to-LDS int4, per the builtin's signature
// (diagnostic showed param 1 as "int4 __device__ *", i.e. addrspace(1) int4*).
typedef __attribute__((address_space(1))) i32x4* gptr128;
typedef __attribute__((address_space(3))) i32x4* lptr128;

#define LDSS 40  // LDS row stride in bf16 (32 data + 8 pad; keeps 16B alignment)

#if defined(__has_builtin)
#if __has_builtin(__builtin_amdgcn_global_load_async_to_lds_b128)
#define HAVE_ASYNC 1
#else
#define HAVE_ASYNC 0
#endif
#else
#define HAVE_ASYNC 0
#endif

__device__ __forceinline__ bf16x16 cat8(bf16x8 a, bf16x8 b) {
  return __builtin_shufflevector(a, b, 0, 1, 2, 3, 4, 5, 6, 7,
                                 8, 9, 10, 11, 12, 13, 14, 15);
}

// ---------------------------------------------------------------------------
// Register-prefetch staging of a ROWSx32 tile (rows x k) as bf16.
// 256 threads; fp32 sources are converted to bf16 at load time so the
// in-flight register footprint stays small.
// ---------------------------------------------------------------------------
template <int ROWS, bool SRC_F32>
struct Fetch {
  static constexpr int TPR = 256 / ROWS;  // threads per row
  static constexpr int EPT = 32 / TPR;    // elements per thread
  static constexpr int NV = EPT / 8;      // bf16x8 chunks per thread
  bf16x8 h[NV];

  __device__ __forceinline__ void load(const void* __restrict__ src, long row0,
                                       int k0, int ld, int tid) {
    const int r = tid / TPR;
    const int koff = (tid % TPR) * EPT;
    if (SRC_F32) {
      const float* s = (const float*)src + (row0 + r) * (long)ld + k0 + koff;
#pragma unroll
      for (int i = 0; i < NV; ++i) {
        f32x4 a = ((const f32x4*)s)[2 * i];
        f32x4 b = ((const f32x4*)s)[2 * i + 1];
        bf16x8 o;
#pragma unroll
        for (int j = 0; j < 4; ++j) {
          o[j] = (bf16)a[j];
          o[4 + j] = (bf16)b[j];
        }
        h[i] = o;
      }
    } else {
      const bf16* s = (const bf16*)src + (row0 + r) * (long)ld + k0 + koff;
#pragma unroll
      for (int i = 0; i < NV; ++i) h[i] = ((const bf16x8*)s)[i];
    }
  }

  __device__ __forceinline__ void store(bf16* __restrict__ dst, int tid) {
    const int r = tid / TPR;
    const int koff = (tid % TPR) * EPT;
    bf16x8* d = (bf16x8*)(dst + r * LDSS + koff);
#pragma unroll
    for (int i = 0; i < NV; ++i) d[i] = h[i];
  }
};

// ---------------------------------------------------------------------------
// Async global->LDS staging (bf16 sources only): GLOBAL_LOAD_ASYNC_TO_LDS_B128,
// tracked by ASYNCcnt -- no VGPR round trip, overlapped with WMMA compute.
// ---------------------------------------------------------------------------
template <int ROWS>
__device__ __forceinline__ void stage_async(const void* __restrict__ src,
                                            long row0, int k0, int ld,
                                            bf16* __restrict__ dst, int tid) {
#if HAVE_ASYNC
  constexpr int TPR = 256 / ROWS;
  constexpr int EPT = 32 / TPR;
  constexpr int NV = EPT / 8;
  const int r = tid / TPR;
  const int koff = (tid % TPR) * EPT;
  const bf16* s = (const bf16*)src + (row0 + r) * (long)ld + k0 + koff;
  bf16* d = dst + r * LDSS + koff;
#pragma unroll
  for (int i = 0; i < NV; ++i) {
    __builtin_amdgcn_global_load_async_to_lds_b128(
        (gptr128)(s + 8 * i), (lptr128)(d + 8 * i), 0, 0);
  }
#else
  (void)src; (void)row0; (void)k0; (void)ld; (void)dst; (void)tid;
#endif
}

__device__ __forceinline__ void wait_async_lds() {
#if HAVE_ASYNC
  asm volatile("s_wait_asynccnt 0x0" ::: "memory");
#endif
}

// ---------------------------------------------------------------------------
// Generic bf16 WMMA GEMM:  C[m,n] = (sum_k A[m,k] * B[n,k] + bias[n]) * scale
//   A: [M,K] row-major (fp32 or bf16), row stride lda
//   B: [N,K] row-major (fp32 or bf16), row stride ldb    (i.e. C = A * B^T)
//   C: bf16 or fp32; normal store C[m*ldc+n] or transposed C[n*ldc+m]
//   grid.z = batch; sAb/sBb/sCb per-batch element strides
// Block tile 256x128, BK=32, double-buffered LDS. 8 waves = 4(M) x 2(N),
// wave tile 64x64 -> 16 v_wmma_f32_16x16x32_bf16 per K-step per wave.
// ---------------------------------------------------------------------------
template <bool A_F32, bool B_F32, bool OUT_F32, bool TRANS_OUT, bool BIAS>
__global__ __launch_bounds__(256) void gemm_bf16_wmma(
    const void* __restrict__ Ag, const void* __restrict__ Bg,
    const float* __restrict__ bias, void* __restrict__ Cg,
    int N, int K, int lda, int ldb, int ldc,
    long sAb, long sBb, long sCb, float scale) {
  __shared__ bf16 As[2][256 * LDSS];  // 40 KB
  __shared__ bf16 Bs[2][128 * LDSS];  // 20 KB

  const int tid = threadIdx.x;
  const int lane = tid & 31;
  const int wave = tid >> 5;
  const int wm = wave >> 1;    // 0..3 -> 64-row slice of M
  const int wn = wave & 1;     // 0..1 -> 64-col slice of N
  const int half = lane >> 4;  // 0..1
  const int l16 = lane & 15;

  const long row0 = (long)blockIdx.y * 256;
  const long col0 = (long)blockIdx.x * 128;

  const void* Abase = A_F32 ? (const void*)((const float*)Ag + blockIdx.z * sAb)
                            : (const void*)((const bf16*)Ag + blockIdx.z * sAb);
  const void* Bbase = B_F32 ? (const void*)((const float*)Bg + blockIdx.z * sBb)
                            : (const void*)((const bf16*)Bg + blockIdx.z * sBb);
  float* Cf = (float*)Cg + blockIdx.z * sCb;
  bf16* Cb = (bf16*)Cg + blockIdx.z * sCb;

  const bool ASY_A = HAVE_ASYNC && !A_F32;
  const bool ASY_B = HAVE_ASYNC && !B_F32;

  f32x8 acc[4][4];
#pragma unroll
  for (int mt = 0; mt < 4; ++mt)
#pragma unroll
    for (int nt = 0; nt < 4; ++nt) acc[mt][nt] = (f32x8)0.0f;

  Fetch<256, A_F32> fa;
  Fetch<128, B_F32> fb;

  // Prologue: stage k=0 into buffer 0.
  if (ASY_A) {
    stage_async<256>(Abase, row0, 0, lda, As[0], tid);
  } else {
    fa.load(Abase, row0, 0, lda, tid);
    fa.store(As[0], tid);
  }
  if (ASY_B) {
    stage_async<128>(Bbase, col0, 0, ldb, Bs[0], tid);
  } else {
    fb.load(Bbase, col0, 0, ldb, tid);
    fb.store(Bs[0], tid);
  }
  wait_async_lds();
  __syncthreads();

  int cur = 0;
  for (int k0 = 0; k0 < K; k0 += 32) {
    const bool have_next = (k0 + 32) < K;
    const int nxt = cur ^ 1;

    // Kick off next tile's fetch before touching this tile's math.
    if (have_next) {
      if (ASY_A) stage_async<256>(Abase, row0, k0 + 32, lda, As[nxt], tid);
      else fa.load(Abase, row0, k0 + 32, lda, tid);
      if (ASY_B) stage_async<128>(Bbase, col0, k0 + 32, ldb, Bs[nxt], tid);
      else fb.load(Bbase, col0, k0 + 32, ldb, tid);
    }

    // B fragments: lane holds col n = base+l16, k = half*16 .. half*16+15
    bf16x16 bfrag[4];
#pragma unroll
    for (int nt = 0; nt < 4; ++nt) {
      const bf16* p = &Bs[cur][(wn * 64 + nt * 16 + l16) * LDSS + half * 16];
      bfrag[nt] = cat8(((const bf16x8*)p)[0], ((const bf16x8*)p)[1]);
    }
    // A fragments: lane holds row m = base+l16, k = half*16 .. half*16+15
#pragma unroll
    for (int mt = 0; mt < 4; ++mt) {
      const bf16* p = &As[cur][(wm * 64 + mt * 16 + l16) * LDSS];
      bf16x16 afrag = cat8(*(const bf16x8*)(p + half * 16),
                           *(const bf16x8*)(p + half * 16 + 8));
#pragma unroll
      for (int nt = 0; nt < 4; ++nt) {
        acc[mt][nt] = __builtin_amdgcn_wmma_f32_16x16x32_bf16(
            false, afrag, false, bfrag[nt], (short)0, acc[mt][nt], false,
            false);
      }
    }

    if (have_next) {
      if (!ASY_A) fa.store(As[nxt], tid);
      if (!ASY_B) fb.store(Bs[nxt], tid);
    }
    wait_async_lds();
    __syncthreads();
    cur = nxt;
  }

  // C/D layout: lane (half,l16), VGPR v -> row m = half*8+v, col n = l16
#pragma unroll
  for (int mt = 0; mt < 4; ++mt) {
#pragma unroll
    for (int nt = 0; nt < 4; ++nt) {
      const long n = col0 + wn * 64 + nt * 16 + l16;
      const float bv = BIAS ? bias[n] : 0.0f;
#pragma unroll
      for (int v = 0; v < 8; ++v) {
        const long m = row0 + wm * 64 + mt * 16 + half * 8 + v;
        const float val = (acc[mt][nt][v] + bv) * scale;
        const long idx = TRANS_OUT ? (n * (long)ldc + m) : (m * (long)ldc + n);
        if (OUT_F32) Cf[idx] = val;
        else         Cb[idx] = (bf16)val;
      }
    }
  }
  (void)N;
}

// ---------------------------------------------------------------------------
// Row-wise softmax over bf16 scores, in place. One 256-thread block per row,
// S == 2048 -> 8 elements per thread. fp32 max/sum via LDS tree reduction.
// ---------------------------------------------------------------------------
__global__ __launch_bounds__(256) void softmax_rows_bf16(bf16* __restrict__ P,
                                                         int S) {
  __shared__ float red[256];
  const long row = blockIdx.x;
  bf16* p = P + row * (long)S;
  const int t = threadIdx.x;

  bf16x8 xv = ((bf16x8*)p)[t];
  float x[8];
#pragma unroll
  for (int i = 0; i < 8; ++i) x[i] = (float)xv[i];

  float mx = x[0];
#pragma unroll
  for (int i = 1; i < 8; ++i) mx = fmaxf(mx, x[i]);
  red[t] = mx;
  __syncthreads();
  for (int s = 128; s > 0; s >>= 1) {
    if (t < s) red[t] = fmaxf(red[t], red[t + s]);
    __syncthreads();
  }
  mx = red[0];
  __syncthreads();

  float sum = 0.0f;
#pragma unroll
  for (int i = 0; i < 8; ++i) {
    x[i] = expf(x[i] - mx);
    sum += x[i];
  }
  red[t] = sum;
  __syncthreads();
  for (int s = 128; s > 0; s >>= 1) {
    if (t < s) red[t] += red[t + s];
    __syncthreads();
  }
  const float inv = 1.0f / red[0];

  bf16x8 ov;
#pragma unroll
  for (int i = 0; i < 8; ++i) ov[i] = (bf16)(x[i] * inv);
  ((bf16x8*)p)[t] = ov;
}

// ---------------------------------------------------------------------------
// Launch: q/k/v proj (fp32->bf16) -> scores -> softmax -> P*V -> out proj
// Workspace (bf16): Qb | Kb | Vt(transposed [e, b*S+s]) | P | Ob   (~96 MB)
// ---------------------------------------------------------------------------
extern "C" void kernel_launch(void* const* d_in, const int* in_sizes, int n_in,
                              void* d_out, int out_size, void* d_ws,
                              size_t ws_size, hipStream_t stream) {
  (void)in_sizes; (void)n_in; (void)out_size; (void)ws_size;
  const int B = 4, S = 2048, D = 1024;
  const int MBS = B * S;  // 8192

  const float* q  = (const float*)d_in[0];
  const float* k  = (const float*)d_in[1];
  const float* v  = (const float*)d_in[2];
  const float* Wq = (const float*)d_in[3];
  const float* bq = (const float*)d_in[4];
  const float* Wk = (const float*)d_in[5];
  const float* bk = (const float*)d_in[6];
  const float* Wv = (const float*)d_in[7];
  const float* bv = (const float*)d_in[8];
  const float* Wo = (const float*)d_in[9];
  const float* bo = (const float*)d_in[10];
  float* out = (float*)d_out;

  const size_t BSD = (size_t)B * S * D;  // 8388608
  const size_t BSS = (size_t)B * S * S;  // 16777216
  bf16* Qb = (bf16*)d_ws;
  bf16* Kb = Qb + BSD;
  bf16* Vt = Kb + BSD;  // [D, B*S]
  bf16* P  = Vt + BSD;  // [B, S, S]
  bf16* Ob = P + BSS;   // [B, S, D]

  const dim3 blk(256);
  const float qscale = 0.03125f;  // 1/sqrt(1024), folded into Q projection

  // Q = (query @ Wq^T + bq) * qscale        -> Qb [b*S, D] bf16
  gemm_bf16_wmma<true, true, false, false, true>
      <<<dim3(D / 128, MBS / 256, 1), blk, 0, stream>>>(
          q, Wq, bq, Qb, D, D, D, D, D, 0, 0, 0, qscale);
  // K = key @ Wk^T + bk                     -> Kb [b*S, D] bf16
  gemm_bf16_wmma<true, true, false, false, true>
      <<<dim3(D / 128, MBS / 256, 1), blk, 0, stream>>>(
          k, Wk, bk, Kb, D, D, D, D, D, 0, 0, 0, 1.0f);
  // V = value @ Wv^T + bv, stored transposed -> Vt [D, b*S] bf16
  gemm_bf16_wmma<true, true, false, true, true>
      <<<dim3(D / 128, MBS / 256, 1), blk, 0, stream>>>(
          v, Wv, bv, Vt, D, D, D, D, MBS, 0, 0, 0, 1.0f);
  // scores[b] = Qb[b] @ Kb[b]^T             -> P [b, S, S] bf16
  gemm_bf16_wmma<false, false, false, false, false>
      <<<dim3(S / 128, S / 256, B), blk, 0, stream>>>(
          Qb, Kb, nullptr, P, S, D, D, D, S,
          (long)S * D, (long)S * D, (long)S * S, 1.0f);
  // softmax rows of P (in place)
  softmax_rows_bf16<<<dim3(MBS), blk, 0, stream>>>(P, S);
  // Ob[b] = P[b] @ V[b]   (B operand = Vt rows, row stride MBS)
  gemm_bf16_wmma<false, false, false, false, false>
      <<<dim3(D / 128, S / 256, B), blk, 0, stream>>>(
          P, Vt, nullptr, Ob, D, S, S, MBS, D,
          (long)S * S, (long)S, (long)S * D, 1.0f);
  // out = Ob @ Wo^T + bo                    -> fp32 output
  gemm_bf16_wmma<false, true, true, false, true>
      <<<dim3(D / 128, MBS / 256, 1), blk, 0, stream>>>(
          Ob, Wo, bo, out, D, D, D, D, D, 0, 0, 0, 1.0f);
}
